// AttentionDirectedBipartiteMessagePassing_48747878809763
// MI455X (gfx1250) — matrix operand order repaired and checked by reading, not compiled, not run on hardware
//
#include <hip/hip_runtime.h>
#include <hip/hip_bf16.h>

typedef __bf16 bf16;
typedef __attribute__((ext_vector_type(16))) __bf16 v16bf;
typedef __attribute__((ext_vector_type(8)))  float  v8f;
typedef __attribute__((ext_vector_type(4)))  unsigned int v4u;

union BFU { v16bf v; v4u u[2]; };

#define XT_S 168   // LDS row stride for the 160-wide x tile (bf16 elems)
#define H_S  136   // LDS row stride for 128-wide hidden tiles

__device__ __forceinline__ v8f wmma_bf16(v16bf a, v16bf b, v8f c) {
  return __builtin_amdgcn_wmma_f32_16x16x32_bf16(false, a, false, b, (short)0, c,
                                                 false, false);
}

// A fragment (16x32 bf16) from a row-major tile (LDS): lane&15 = row,
// lane-half selects K {+0..7, +16..23} vs {+8..15, +24..31}.
__device__ __forceinline__ v16bf load_a(const bf16* base, int stride, int m0, int k0, int lane) {
  int m  = m0 + (lane & 15);
  int kb = k0 + ((lane >> 4) << 3);
  const bf16* p = base + m * stride + kb;
  BFU r;
  r.u[0] = *(const v4u*)p;
  r.u[1] = *(const v4u*)(p + 16);
  return r.v;
}

// B fragment (32x16 bf16) from transposed weights Wt[N][K] row-major:
// lane&15 = column n, lane-half selects K {+0..7, +16..23} vs {+8..15, +24..31}.
__device__ __forceinline__ v16bf load_b(const bf16* wt, int K, int n0, int k0, int lane) {
  int n  = n0 + (lane & 15);
  int kb = k0 + ((lane >> 4) << 3);
  const bf16* p = wt + n * K + kb;
  BFU r;
  r.u[0] = *(const v4u*)p;
  r.u[1] = *(const v4u*)(p + 16);
  return r.v;
}

__device__ __forceinline__ void atomic_max_f32(float* addr, float val) {
  if (val >= 0.f) atomicMax((int*)addr, __float_as_int(val));
  else            atomicMin((unsigned int*)addr, __float_as_uint(val));
}

// ---------------------------------------------------------------- init
__global__ void init_kernel(float* aggr, float* segmax, float* denom, int Nd) {
  int t = blockIdx.x * blockDim.x + threadIdx.x;
  if (t < Nd * 128) aggr[t] = 0.f;
  if (t < Nd * 8) { segmax[t] = -__builtin_inff(); denom[t] = 0.f; }
}

// ------------------------------------------------- weight transpose->bf16
__global__ void prep_kernel(const float* kW0, const float* kW1, const float* vW0,
                            const float* vW1, const float* oW0, const float* oW1,
                            bf16* kW0t, bf16* kW1t, bf16* vW0t, bf16* vW1t,
                            bf16* oW0t, bf16* oW1t) {
  int t = blockIdx.x * 256 + threadIdx.x;
  if (t < 160 * 128) {
    int k = t >> 7, n = t & 127;
    kW0t[n * 160 + k] = (bf16)kW0[t];
    vW0t[n * 160 + k] = (bf16)vW0[t];
  }
  if (t < 128 * 128) {
    int k = t >> 7, n = t & 127;
    kW1t[n * 128 + k] = (bf16)kW1[t];
    vW1t[n * 128 + k] = (bf16)vW1[t];
    oW0t[n * 128 + k] = (bf16)oW0[t];
    oW1t[n * 128 + k] = (bf16)oW1[t];
  }
}

// --------------------------------- fused per-edge K/V MLP + q.k + segmax
__global__ __launch_bounds__(256) void edge_kv_kernel(
    const float* __restrict__ xs, const float* __restrict__ xd,
    const float* __restrict__ ea, const int* __restrict__ ei,
    const float* __restrict__ q,
    const float* __restrict__ kb0, const float* __restrict__ kb1,
    const float* __restrict__ vb0, const float* __restrict__ vb1,
    const bf16* __restrict__ kW0t, const bf16* __restrict__ kW1t,
    const bf16* __restrict__ vW0t, const bf16* __restrict__ vW1t,
    float* __restrict__ cbuf, bf16* __restrict__ vws,
    float* __restrict__ segmax, int E) {
  __shared__ bf16 xt[64 * XT_S];
  __shared__ bf16 h0[2][64 * H_S];
  const int tid = threadIdx.x, lane = tid & 31, wave = tid >> 5;
  const int e0 = blockIdx.x * 64;
  const int* src = ei;
  const int* dst = ei + E;

  { // stage concat(x_src, x_dst, edge_attr) tile as bf16 (4 threads / row)
    int r = tid >> 2;
    int e = e0 + r;
    bool ok = e < E;
    int ec = ok ? e : 0;
    int s = src[ec], d = dst[ec];
    for (int c = tid & 3; c < 160; c += 4) {
      float v;
      if (c < 64)       v = xs[s * 64 + c];
      else if (c < 128) v = xd[d * 64 + (c - 64)];
      else              v = ea[(size_t)ec * 32 + (c - 128)];
      xt[r * XT_S + c] = (bf16)(ok ? v : 0.f);
    }
  }
  __syncthreads();

  const int mat  = wave >> 2;   // 0 = K path, 1 = V path
  const int ncol = wave & 3;    // owns n-tiles {ncol, ncol+4}
  const bf16*  W0t = mat ? vW0t : kW0t;
  const bf16*  W1t = mat ? vW1t : kW1t;
  const float* b0  = mat ? vb0 : kb0;
  const float* b1  = mat ? vb1 : kb1;
  bf16* h = h0[mat];

  // layer 0: [64x160] @ [160x128], ReLU, no residual (non-square)
  {
    v8f acc[2][4];
#pragma unroll
    for (int t = 0; t < 2; ++t)
#pragma unroll
      for (int ms = 0; ms < 4; ++ms) acc[t][ms] = {};
#pragma unroll
    for (int ks = 0; ks < 5; ++ks) {
      v16bf bf0 = load_b(W0t, 160, ncol * 16, ks * 32, lane);
      v16bf bf1 = load_b(W0t, 160, (ncol + 4) * 16, ks * 32, lane);
#pragma unroll
      for (int ms = 0; ms < 4; ++ms) {
        v16bf a = load_a(xt, XT_S, ms * 16, ks * 32, lane);
        acc[0][ms] = wmma_bf16(a, bf0, acc[0][ms]);
        acc[1][ms] = wmma_bf16(a, bf1, acc[1][ms]);
      }
    }
#pragma unroll
    for (int t = 0; t < 2; ++t) {
      int nb = (ncol + 4 * t) * 16;
      float bv = b0[nb + (lane & 15)];
#pragma unroll
      for (int ms = 0; ms < 4; ++ms)
#pragma unroll
        for (int r = 0; r < 8; ++r) {
          int m = ms * 16 + r + ((lane >> 4) << 3);
          float z = acc[t][ms][r] + bv;
          h[m * H_S + nb + (lane & 15)] = (bf16)fmaxf(z, 0.f);
        }
    }
  }
  __syncthreads();

  // layer 1: [64x128] @ [128x128] + residual; K path -> q.k; V path -> store
  {
    v8f acc[2][4];
#pragma unroll
    for (int t = 0; t < 2; ++t)
#pragma unroll
      for (int ms = 0; ms < 4; ++ms) acc[t][ms] = {};
#pragma unroll
    for (int ks = 0; ks < 4; ++ks) {
      v16bf bf0 = load_b(W1t, 128, ncol * 16, ks * 32, lane);
      v16bf bf1 = load_b(W1t, 128, (ncol + 4) * 16, ks * 32, lane);
#pragma unroll
      for (int ms = 0; ms < 4; ++ms) {
        v16bf a = load_a(h, H_S, ms * 16, ks * 32, lane);
        acc[0][ms] = wmma_bf16(a, bf0, acc[0][ms]);
        acc[1][ms] = wmma_bf16(a, bf1, acc[1][ms]);
      }
    }
#pragma unroll
    for (int t = 0; t < 2; ++t) {
      int nt = ncol + 4 * t;        // n-tile index == attention head
      int nb = nt * 16;
      int nl = lane & 15;
      float bv = b1[nb + nl];
      float qv = q[nb + nl];
#pragma unroll
      for (int ms = 0; ms < 4; ++ms) {
#pragma unroll
        for (int r = 0; r < 8; ++r) {
          int m = ms * 16 + r + ((lane >> 4) << 3);
          int e = e0 + m;
          float z = acc[t][ms][r] + bv + (float)h[m * H_S + nb + nl];
          if (mat) {
            if (e < E) vws[(size_t)e * 128 + nb + nl] = (bf16)z;
          } else {
            float p = z * qv;                 // q . k over this head's 16 cols
            p += __shfl_xor(p, 8, 32);
            p += __shfl_xor(p, 4, 32);
            p += __shfl_xor(p, 2, 32);
            p += __shfl_xor(p, 1, 32);
            if (nl == 0 && e < E) {
              float c = p * 0.25f;            // 1/sqrt(HD=16)
              cbuf[(size_t)e * 8 + nt] = c;
              atomic_max_f32(&segmax[(size_t)dst[e] * 8 + nt], c);
            }
          }
        }
      }
    }
  }
}

// ----------------------------------------------- softmax exp + denominator
__global__ void softmax_exp_kernel(float* cbuf, const float* __restrict__ segmax,
                                   float* denom, const int* __restrict__ dst, int E) {
  int t = blockIdx.x * 256 + threadIdx.x;
  if (t >= E * 8) return;
  int e = t >> 3, hh = t & 7;
  int d = dst[e];
  float ex = __expf(cbuf[t] - segmax[d * 8 + hh]);
  cbuf[t] = ex;
  atomicAdd(&denom[d * 8 + hh], ex);
}

// -------------------------------------- weighted message scatter into aggr
__global__ __launch_bounds__(256) void scatter_kernel(
    const float* __restrict__ cbuf, const float* __restrict__ denom,
    const bf16* __restrict__ vws, const int* __restrict__ dst,
    float* __restrict__ aggr, int E) {
  int tid = threadIdx.x;
  int e = blockIdx.x * 2 + (tid >> 7);
  if (e >= E) return;
  int c = tid & 127, hh = c >> 4;
  int d = dst[e];
  float w = cbuf[(size_t)e * 8 + hh] / (denom[d * 8 + hh] + 1e-16f);
  float val = w * (float)vws[(size_t)e * 128 + c];
  atomicAdd(&aggr[(size_t)d * 128 + c], val);
}

// -------------------------------- per-node output residual stack (in place)
__global__ __launch_bounds__(256) void out_kernel(
    float* __restrict__ io, const bf16* __restrict__ oW0t,
    const bf16* __restrict__ oW1t, const float* __restrict__ ob0,
    const float* __restrict__ ob1, int N) {
  __shared__ bf16 at[64 * H_S];
  __shared__ bf16 h1[64 * H_S];
  const int tid = threadIdx.x, lane = tid & 31, wave = tid >> 5;
  const int n0 = blockIdx.x * 64;
  {
    int r = tid >> 2;
    int e = n0 + r;
    bool ok = e < N;
    for (int c = tid & 3; c < 128; c += 4) {
      float v = ok ? io[(size_t)e * 128 + c] : 0.f;
      at[r * H_S + c] = (bf16)fmaxf(v, 0.f);   // relu(aggr)
    }
  }
  __syncthreads();
  const int nb = wave * 16;
  const int nl = lane & 15;
  {
    float bv = ob0[nb + nl];
    v8f acc[4];
#pragma unroll
    for (int ms = 0; ms < 4; ++ms) acc[ms] = {};
#pragma unroll
    for (int ks = 0; ks < 4; ++ks) {
      v16bf b = load_b(oW0t, 128, nb, ks * 32, lane);
#pragma unroll
      for (int ms = 0; ms < 4; ++ms) {
        v16bf a = load_a(at, H_S, ms * 16, ks * 32, lane);
        acc[ms] = wmma_bf16(a, b, acc[ms]);
      }
    }
#pragma unroll
    for (int ms = 0; ms < 4; ++ms)
#pragma unroll
      for (int r = 0; r < 8; ++r) {
        int m = ms * 16 + r + ((lane >> 4) << 3);
        float z = acc[ms][r] + bv + (float)at[m * H_S + nb + nl];
        h1[m * H_S + nb + nl] = (bf16)fmaxf(z, 0.f);
      }
  }
  __syncthreads();
  {
    float bv = ob1[nb + nl];
    v8f acc[4];
#pragma unroll
    for (int ms = 0; ms < 4; ++ms) acc[ms] = {};
#pragma unroll
    for (int ks = 0; ks < 4; ++ks) {
      v16bf b = load_b(oW1t, 128, nb, ks * 32, lane);
#pragma unroll
      for (int ms = 0; ms < 4; ++ms) {
        v16bf a = load_a(h1, H_S, ms * 16, ks * 32, lane);
        acc[ms] = wmma_bf16(a, b, acc[ms]);
      }
    }
#pragma unroll
    for (int ms = 0; ms < 4; ++ms)
#pragma unroll
      for (int r = 0; r < 8; ++r) {
        int m = ms * 16 + r + ((lane >> 4) << 3);
        int e = n0 + m;
        if (e < N) {
          float z = acc[ms][r] + bv + (float)h1[m * H_S + nb + nl];
          io[(size_t)e * 128 + nb + nl] = fmaxf(z, 0.f);
        }
      }
  }
}

extern "C" void kernel_launch(void* const* d_in, const int* in_sizes, int n_in,
                              void* d_out, int out_size, void* d_ws, size_t ws_size,
                              hipStream_t stream) {
  const float* x_src = (const float*)d_in[0];
  const float* x_dst = (const float*)d_in[1];
  const float* ea    = (const float*)d_in[2];
  const int*   ei    = (const int*)d_in[3];
  const float* q     = (const float*)d_in[4];
  const float* kW0 = (const float*)d_in[5];  const float* kb0 = (const float*)d_in[6];
  const float* kW1 = (const float*)d_in[7];  const float* kb1 = (const float*)d_in[8];
  const float* vW0 = (const float*)d_in[9];  const float* vb0 = (const float*)d_in[10];
  const float* vW1 = (const float*)d_in[11]; const float* vb1 = (const float*)d_in[12];
  const float* oW0 = (const float*)d_in[13]; const float* ob0 = (const float*)d_in[14];
  const float* oW1 = (const float*)d_in[15]; const float* ob1 = (const float*)d_in[16];

  const int E  = in_sizes[2] / 32;   // edge_attr is [E,32]
  const int Nd = in_sizes[1] / 64;   // x_dst is [N_DST,64]

  char* ws = (char*)d_ws;
  bf16* kW0t = (bf16*)ws;                  // 160*128
  bf16* vW0t = kW0t + 160 * 128;
  bf16* kW1t = vW0t + 160 * 128;           // 128*128 each below
  bf16* vW1t = kW1t + 128 * 128;
  bf16* oW0t = vW1t + 128 * 128;
  bf16* oW1t = oW0t + 128 * 128;
  // weights end at 212992 bytes (256-aligned)
  float* segmax = (float*)(ws + 212992);
  float* denom  = segmax + (size_t)Nd * 8;
  float* cbuf   = denom + (size_t)Nd * 8;
  bf16*  vws    = (bf16*)(cbuf + (size_t)E * 8);
  float* out    = (float*)d_out;           // doubles as aggr accumulator

  init_kernel<<<(Nd * 128 + 255) / 256, 256, 0, stream>>>(out, segmax, denom, Nd);
  prep_kernel<<<(160 * 128 + 255) / 256, 256, 0, stream>>>(
      kW0, kW1, vW0, vW1, oW0, oW1, kW0t, kW1t, vW0t, vW1t, oW0t, oW1t);
  edge_kv_kernel<<<(E + 63) / 64, 256, 0, stream>>>(
      x_src, x_dst, ea, ei, q, kb0, kb1, vb0, vb1,
      kW0t, kW1t, vW0t, vW1t, cbuf, vws, segmax, E);
  softmax_exp_kernel<<<(E * 8 + 255) / 256, 256, 0, stream>>>(cbuf, segmax, denom,
                                                              ei + E, E);
  scatter_kernel<<<(E + 1) / 2, 256, 0, stream>>>(cbuf, denom, vws, ei + E, out, E);
  out_kernel<<<(Nd + 63) / 64, 256, 0, stream>>>(out, oW0t, oW1t, ob0, ob1, Nd);
}
